// LSTMs_43173011259785
// MI455X (gfx1250) — matrix operand order repaired and checked by reading, not compiled, not run on hardware
//
#include <hip/hip_runtime.h>

#define B_    64
#define P_    196
#define EE_   2048
#define A_    512
#define D_    512
#define V_    10000
#define VP_   10048  // V_ padded so NP/16 = 628 n-tiles (divisible by 4)
#define TCAP_ 21
#define TDEC_ 20
#define EMB_  512
#define X_    2560   // EMB_ + EE_
#define G4_   2048   // 4*D_

typedef __attribute__((ext_vector_type(16))) __bf16 v16bf;
typedef __attribute__((ext_vector_type(8)))  float  v8f;

__device__ __forceinline__ float sigf(float x) { return 1.0f / (1.0f + expf(-x)); }

// ---------------------------------------------------------------------------
// WMMA GEMM: Out(M,N) = A(M,K) * W(N,K)^T + bias(N) [+ Cin(M,N)]
// A fp32 (converted to bf16 on the fly), W bf16.
// One 16-row M-tile per block; each wave owns 4 consecutive 16-col N-tiles:
// the A fragment is converted once per K-step and feeds 4 independent WMMAs.
// All fragment loads for a K-step are issued before the WMMA chain so the
// scheduler can clause them and overlap next-iteration loads with WMMAs.
// Requires: M % 16 == 0, N % 64 == 0, K % 32 == 0 (all call sites satisfy).
// rowmap (optional) indirects A rows (argsort-reordered encoder).
// ---------------------------------------------------------------------------
__global__ __launch_bounds__(128)
void gemm_wmma_bf16(const float* __restrict__ Amat, const int* __restrict__ rowmap,
                    const __bf16* __restrict__ W, const float* __restrict__ bias,
                    const float* __restrict__ Cin, float* __restrict__ Out,
                    int M, int N, int K) {
    // readfirstlane -> SGPR: keeps all tile indexing scalar so no EXEC masking
    // is ever generated around the WMMAs (ISA requires EXEC all-ones).
    const int wave = __builtin_amdgcn_readfirstlane(threadIdx.x >> 5);
    const int lane = threadIdx.x & 31;
    const int half = lane >> 4;
    const int r    = lane & 15;

    const int mtile  = blockIdx.y;
    const int ntiles = N >> 4;
    const int ntile0 = blockIdx.x * 16 + wave * 4;
    if (ntile0 >= ntiles) return;               // scalar branch (tail blocks)

    const int mrow = mtile * 16 + r;
    const int arow = rowmap ? rowmap[mrow] : mrow;
    const float* Ap = Amat + (size_t)arow * K;

    const __bf16* Wp[4];
#pragma unroll
    for (int j = 0; j < 4; ++j)
        Wp[j] = W + (size_t)((ntile0 + j) * 16 + r) * K;

    v8f acc[4];
    if (Cin) {
#pragma unroll
        for (int j = 0; j < 4; ++j) {
            const float* cp = Cin + (size_t)(mtile * 16 + 8 * half) * N + (ntile0 + j) * 16 + r;
#pragma unroll
            for (int e = 0; e < 8; ++e) acc[j][e] = cp[(size_t)e * N];
        }
    } else {
#pragma unroll
        for (int j = 0; j < 4; ++j)
#pragma unroll
            for (int e = 0; e < 8; ++e) acc[j][e] = 0.0f;
    }

    for (int k0 = 0; k0 < K; k0 += 32) {
        // Issue ALL loads for this K-step first (clause-friendly, hoistable).
        // B fragments (B 32x16, contiguous k order used by this code): lane half h holds contiguous K [16h,16h+16)
        v16bf bfr[4];
#pragma unroll
        for (int j = 0; j < 4; ++j)
            bfr[j] = *(const v16bf*)(Wp[j] + k0 + 16 * half);
        // A fragment: lane half h holds contiguous K [16h,16h+16) (matches B)
        v8f alo = *(const v8f*)(Ap + k0 + 16 * half);
        v8f ahi = *(const v8f*)(Ap + k0 + 16 * half + 8);
        v16bf afrag;
#pragma unroll
        for (int i = 0; i < 8; ++i) {
            afrag[i]     = (__bf16)alo[i];
            afrag[8 + i] = (__bf16)ahi[i];
        }
        // 4 independent WMMAs sharing the A fragment.
#pragma unroll
        for (int j = 0; j < 4; ++j)
            acc[j] = __builtin_amdgcn_wmma_f32_16x16x32_bf16(
                false, afrag, false, bfr[j], (short)0, acc[j], false, false);
    }

#pragma unroll
    for (int j = 0; j < 4; ++j) {
        const int ncol = (ntile0 + j) * 16 + r;
        const float bv = bias ? bias[ncol] : 0.0f;
        float* op = Out + (size_t)(mtile * 16 + 8 * half) * N + ncol;
#pragma unroll
        for (int e = 0; e < 8; ++e) op[(size_t)e * N] = acc[j][e] + bv;
    }
}

// ---------------------------------------------------------------------------
// Setup kernels
// ---------------------------------------------------------------------------
__global__ void order_kernel(const int* __restrict__ cap_len,
                             int* __restrict__ order, int* __restrict__ dec_len) {
    int b = threadIdx.x;
    if (b >= B_) return;
    int lb = cap_len[b];
    int rank = 0;
    for (int j = 0; j < B_; ++j) {
        int lj = cap_len[j];
        rank += (lj > lb) || (lj == lb && j < b);   // stable argsort(-lens)
    }
    order[rank]   = b;
    dec_len[rank] = lb - 1;
}

__global__ void rowmap_kernel(const int* __restrict__ order, int* __restrict__ rowmap) {
    int idx = blockIdx.x * 256 + threadIdx.x;
    if (idx >= B_ * P_) return;
    rowmap[idx] = order[idx / P_] * P_ + (idx % P_);
}

__global__ void cvt_bf16_kernel(const float* __restrict__ src, __bf16* __restrict__ dst, int n) {
    int idx = blockIdx.x * 256 + threadIdx.x;
    if (idx < n) dst[idx] = (__bf16)src[idx];
}

__global__ void zero_bf16_kernel(__bf16* __restrict__ dst, int n) {
    int idx = blockIdx.x * 256 + threadIdx.x;
    if (idx < n) dst[idx] = (__bf16)0.0f;
}

__global__ void pad_bias_kernel(const float* __restrict__ src, float* __restrict__ dst) {
    int idx = blockIdx.x * 256 + threadIdx.x;
    if (idx < VP_) dst[idx] = (idx < V_) ? src[idx] : 0.0f;
}

__global__ void bias_sum_kernel(const float* __restrict__ a, const float* __restrict__ b,
                                float* __restrict__ out) {
    int idx = blockIdx.x * 256 + threadIdx.x;
    if (idx < G4_) out[idx] = a[idx] + b[idx];
}

__global__ void mean_kernel(const float* __restrict__ enc, const int* __restrict__ order,
                            float* __restrict__ mean) {
    int b = blockIdx.y;
    int e = blockIdx.x * 256 + threadIdx.x;
    const float* ep = enc + (size_t)order[b] * P_ * EE_ + e;
    float s = 0.0f;
    for (int p = 0; p < P_; ++p) s += ep[(size_t)p * EE_];
    mean[b * EE_ + e] = s * (1.0f / (float)P_);
}

// ---------------------------------------------------------------------------
// Per-step kernels
// ---------------------------------------------------------------------------
__global__ __launch_bounds__(256)
void attn_softmax_kernel(const float* __restrict__ enc_att, const float* __restrict__ att2,
                         const float* __restrict__ wfull, const float* __restrict__ bfull,
                         const int* __restrict__ dec_len, int t,
                         float* __restrict__ alpha_ws, float* __restrict__ alphas_out) {
    __shared__ float e_sh[P_];
    __shared__ float red[256];
    const int b    = blockIdx.x;
    const int tid  = threadIdx.x;
    const int wave = tid >> 5, lane = tid & 31;
    const float* a2 = att2 + b * A_;

    for (int p = wave; p < P_; p += 8) {
        const float* ea = enc_att + ((size_t)b * P_ + p) * A_;
        float s = 0.0f;
        for (int a = lane; a < A_; a += 32)
            s += fmaxf(ea[a] + a2[a], 0.0f) * wfull[a];
#pragma unroll
        for (int off = 16; off > 0; off >>= 1) s += __shfl_xor(s, off, 32);
        if (lane == 0) e_sh[p] = s + bfull[0];
    }
    __syncthreads();

    float m = -3.4e38f;
    for (int p = tid; p < P_; p += 256) m = fmaxf(m, e_sh[p]);
    red[tid] = m; __syncthreads();
    for (int s2 = 128; s2 > 0; s2 >>= 1) {
        if (tid < s2) red[tid] = fmaxf(red[tid], red[tid + s2]);
        __syncthreads();
    }
    const float mx = red[0];
    __syncthreads();

    float sum = 0.0f;
    for (int p = tid; p < P_; p += 256) {
        float ex = expf(e_sh[p] - mx);
        e_sh[p] = ex;
        sum += ex;
    }
    red[tid] = sum; __syncthreads();
    for (int s2 = 128; s2 > 0; s2 >>= 1) {
        if (tid < s2) red[tid] += red[tid + s2];
        __syncthreads();
    }
    const float inv = 1.0f / red[0];
    const bool act  = t < dec_len[b];
    for (int p = tid; p < P_; p += 256) {
        float al = e_sh[p] * inv;
        alpha_ws[b * P_ + p] = al;
        alphas_out[((size_t)b * TDEC_ + t) * P_ + p] = act ? al : 0.0f;
    }
}

__global__ void ctx_kernel(const float* __restrict__ alpha, const float* __restrict__ enc,
                           const int* __restrict__ order, float* __restrict__ ctx) {
    int b = blockIdx.y;
    int e = blockIdx.x * 256 + threadIdx.x;
    const float* ep = enc + (size_t)order[b] * P_ * EE_ + e;
    const float* al = alpha + b * P_;
    float s = 0.0f;
    for (int p = 0; p < P_; ++p) s += al[p] * ep[(size_t)p * EE_];
    ctx[b * EE_ + e] = s;
}

__global__ void build_x_kernel(const float* __restrict__ emb_W, const int* __restrict__ cap,
                               const int* __restrict__ order, int t,
                               const float* __restrict__ gate_lin, const float* __restrict__ ctx,
                               float* __restrict__ x) {
    int b = blockIdx.y;
    int j = blockIdx.x * 256 + threadIdx.x;
    float v;
    if (j < EMB_) {
        int tok = cap[order[b] * TCAP_ + t];
        v = emb_W[(size_t)tok * EMB_ + j];
    } else {
        int e = j - EMB_;
        v = sigf(gate_lin[b * EE_ + e]) * ctx[b * EE_ + e];
    }
    x[b * X_ + j] = v;
}

__global__ void lstm_cell_kernel(const float* __restrict__ gates, float* __restrict__ h,
                                 float* __restrict__ c, const int* __restrict__ dec_len, int t) {
    int b = blockIdx.y;
    int j = blockIdx.x * 256 + threadIdx.x;
    const float* g = gates + (size_t)b * G4_;
    float i_g = g[j], f_g = g[D_ + j], g_g = g[2 * D_ + j], o_g = g[3 * D_ + j];
    float cn = sigf(f_g) * c[b * D_ + j] + sigf(i_g) * tanhf(g_g);
    float hn = sigf(o_g) * tanhf(cn);
    if (t < dec_len[b]) {
        c[b * D_ + j] = cn;
        h[b * D_ + j] = hn;
    }
}

__global__ void preds_out_kernel(const float* __restrict__ preds, const int* __restrict__ dec_len,
                                 int t, float* __restrict__ out) {
    int idx = blockIdx.x * 256 + threadIdx.x;
    if (idx >= B_ * V_) return;
    int b = idx / V_, v = idx % V_;
    out[((size_t)b * TDEC_ + t) * V_ + v] = (t < dec_len[b]) ? preds[(size_t)b * VP_ + v] : 0.0f;
}

// ---------------------------------------------------------------------------
// Host launcher
// ---------------------------------------------------------------------------
extern "C" void kernel_launch(void* const* d_in, const int* in_sizes, int n_in,
                              void* d_out, int out_size, void* d_ws, size_t ws_size,
                              hipStream_t stream) {
    const float* encoder_out = (const float*)d_in[0];
    const float* emb_W       = (const float*)d_in[1];
    const float* h_W         = (const float*)d_in[2];
    const float* h_b         = (const float*)d_in[3];
    const float* c_W         = (const float*)d_in[4];
    const float* c_b         = (const float*)d_in[5];
    const float* att_enc_W   = (const float*)d_in[6];
    const float* att_enc_b   = (const float*)d_in[7];
    const float* att_dec_W   = (const float*)d_in[8];
    const float* att_dec_b   = (const float*)d_in[9];
    const float* att_full_W  = (const float*)d_in[10];
    const float* att_full_b  = (const float*)d_in[11];
    const float* sig_W       = (const float*)d_in[12];
    const float* sig_b       = (const float*)d_in[13];
    const float* lstm_Wih    = (const float*)d_in[14];
    const float* lstm_Whh    = (const float*)d_in[15];
    const float* lstm_bih    = (const float*)d_in[16];
    const float* lstm_bhh    = (const float*)d_in[17];
    const float* fc_W        = (const float*)d_in[18];
    const float* fc_b        = (const float*)d_in[19];
    const int*   encoder_cap = (const int*)d_in[20];
    const int*   cap_len     = (const int*)d_in[21];

    float* out_preds  = (float*)d_out;
    float* out_alphas = out_preds + (size_t)B_ * TDEC_ * V_;

    // bump allocator over d_ws
    char*  base = (char*)d_ws;
    size_t off  = 0;
    auto alloc = [&](size_t bytes) -> void* {
        void* p = base + off;
        off = (off + bytes + 255) & ~(size_t)255;
        return p;
    };

    int*    order    = (int*)alloc(B_ * sizeof(int));
    int*    dec_len  = (int*)alloc(B_ * sizeof(int));
    int*    rowmap   = (int*)alloc((size_t)B_ * P_ * sizeof(int));

    __bf16* Wh_bf    = (__bf16*)alloc((size_t)D_ * EE_ * 2);
    __bf16* Wc_bf    = (__bf16*)alloc((size_t)D_ * EE_ * 2);
    __bf16* Wae_bf   = (__bf16*)alloc((size_t)A_ * EE_ * 2);
    __bf16* Wad_bf   = (__bf16*)alloc((size_t)A_ * D_ * 2);
    __bf16* Wsig_bf  = (__bf16*)alloc((size_t)EE_ * D_ * 2);
    __bf16* Wih_bf   = (__bf16*)alloc((size_t)G4_ * X_ * 2);
    __bf16* Whh_bf   = (__bf16*)alloc((size_t)G4_ * D_ * 2);
    __bf16* Wfc_bf   = (__bf16*)alloc((size_t)VP_ * D_ * 2);   // padded rows zeroed

    float* fcb_pad   = (float*)alloc((size_t)VP_ * 4);
    float* mean_enc  = (float*)alloc((size_t)B_ * EE_ * 4);
    float* enc_att   = (float*)alloc((size_t)B_ * P_ * A_ * 4);
    float* h_st      = (float*)alloc((size_t)B_ * D_ * 4);
    float* c_st      = (float*)alloc((size_t)B_ * D_ * 4);
    float* att2      = (float*)alloc((size_t)B_ * A_ * 4);
    float* alpha_ws  = (float*)alloc((size_t)B_ * P_ * 4);
    float* ctx       = (float*)alloc((size_t)B_ * EE_ * 4);
    float* gate_lin  = (float*)alloc((size_t)B_ * EE_ * 4);
    float* x_buf     = (float*)alloc((size_t)B_ * X_ * 4);
    float* gates     = (float*)alloc((size_t)B_ * G4_ * 4);
    float* bias_tot  = (float*)alloc((size_t)G4_ * 4);
    float* preds_ws  = (float*)alloc((size_t)B_ * VP_ * 4);

    // ---- setup ----
    order_kernel<<<1, 64, 0, stream>>>(cap_len, order, dec_len);
    rowmap_kernel<<<(B_ * P_ + 255) / 256, 256, 0, stream>>>(order, rowmap);

    auto cvt = [&](const float* s, __bf16* d, int n) {
        cvt_bf16_kernel<<<(n + 255) / 256, 256, 0, stream>>>(s, d, n);
    };
    cvt(h_W,       Wh_bf,   D_ * EE_);
    cvt(c_W,       Wc_bf,   D_ * EE_);
    cvt(att_enc_W, Wae_bf,  A_ * EE_);
    cvt(att_dec_W, Wad_bf,  A_ * D_);
    cvt(sig_W,     Wsig_bf, EE_ * D_);
    cvt(lstm_Wih,  Wih_bf,  G4_ * X_);
    cvt(lstm_Whh,  Whh_bf,  G4_ * D_);
    cvt(fc_W,      Wfc_bf,  V_ * D_);
    zero_bf16_kernel<<<((VP_ - V_) * D_ + 255) / 256, 256, 0, stream>>>(
        Wfc_bf + (size_t)V_ * D_, (VP_ - V_) * D_);
    pad_bias_kernel<<<(VP_ + 255) / 256, 256, 0, stream>>>(fc_b, fcb_pad);

    bias_sum_kernel<<<G4_ / 256, 256, 0, stream>>>(lstm_bih, lstm_bhh, bias_tot);
    mean_kernel<<<dim3(EE_ / 256, B_), 256, 0, stream>>>(encoder_out, order, mean_enc);

    auto gemm = [&](const float* Amat, const int* rmap, const __bf16* W, const float* bias,
                    const float* Cin, float* Out, int M, int N, int K) {
        const int ntiles = N / 16;
        dim3 grid((ntiles + 15) / 16, M / 16);
        gemm_wmma_bf16<<<grid, 128, 0, stream>>>(Amat, rmap, W, bias, Cin, Out, M, N, K);
    };

    // h0, c0
    gemm(mean_enc, nullptr, Wh_bf, h_b, nullptr, h_st, B_, D_, EE_);
    gemm(mean_enc, nullptr, Wc_bf, c_b, nullptr, c_st, B_, D_, EE_);
    // enc_att = enc(sorted) @ att_enc_W.T + b   (M = B*P = 12544)
    gemm(encoder_out, rowmap, Wae_bf, att_enc_b, nullptr, enc_att, B_ * P_, A_, EE_);

    // ---- decode loop ----
    for (int t = 0; t < TDEC_; ++t) {
        gemm(h_st, nullptr, Wad_bf, att_dec_b, nullptr, att2, B_, A_, D_);
        attn_softmax_kernel<<<B_, 256, 0, stream>>>(enc_att, att2, att_full_W, att_full_b,
                                                    dec_len, t, alpha_ws, out_alphas);
        ctx_kernel<<<dim3(EE_ / 256, B_), 256, 0, stream>>>(alpha_ws, encoder_out, order, ctx);
        gemm(h_st, nullptr, Wsig_bf, sig_b, nullptr, gate_lin, B_, EE_, D_);
        build_x_kernel<<<dim3(X_ / 256, B_), 256, 0, stream>>>(emb_W, encoder_cap, order, t,
                                                               gate_lin, ctx, x_buf);
        gemm(x_buf, nullptr, Wih_bf, bias_tot, nullptr, gates, B_, G4_, X_);
        gemm(h_st,  nullptr, Whh_bf, nullptr,  gates,   gates, B_, G4_, D_);
        lstm_cell_kernel<<<dim3(D_ / 256, B_), 256, 0, stream>>>(gates, h_st, c_st, dec_len, t);
        gemm(h_st, nullptr, Wfc_bf, fcb_pad, nullptr, preds_ws, B_, VP_, D_);
        preds_out_kernel<<<(B_ * V_ + 255) / 256, 256, 0, stream>>>(preds_ws, dec_len, t, out_preds);
    }

    (void)in_sizes; (void)n_in; (void)out_size; (void)ws_size;
}
